// STGAT_3985729651487
// MI455X (gfx1250) — matrix operand order repaired and checked by reading, not compiled, not run on hardware
//
#include <hip/hip_runtime.h>
#include <hip/hip_bf16.h>
#include <math.h>

// ---------------- problem constants (match reference) ----------------
#define NN      3000          // nodes (B=1 -> BN == NN)
#define T_STEPS 12
#define F_INN   2
#define HEADS   8
#define HID     32
#define HC      (HEADS * HID) // 256
#define F_OUTT  2
#define T_OUT   12
#define K_OUT   (F_OUTT * T_OUT)  // 24
#define NEG_SLOPE 0.2f

typedef __attribute__((ext_vector_type(16))) _Float16 v16h;
typedef __attribute__((ext_vector_type(8)))  float    v8f;

// ---------------- helpers ----------------
__device__ __forceinline__ int enc_f32(float f) {
    int b = __float_as_int(f);
    return (b < 0) ? (b ^ 0x7fffffff) : b;       // monotonic float->int map
}
__device__ __forceinline__ float dec_f32(int e) {
    return __int_as_float((e < 0) ? (e ^ 0x7fffffff) : e);
}
// branchless fast sigmoid/tanh: v_exp_f32 + raw v_rcp_f32 (no IEEE div sequence)
__device__ __forceinline__ float sigm_fast(float x) {
    return __builtin_amdgcn_rcpf(1.0f + __expf(-x));
}
__device__ __forceinline__ float tanh_fast(float x) {
    return 1.0f - 2.0f * __builtin_amdgcn_rcpf(__expf(2.0f * x) + 1.0f);
}

// ---------------- 0: re-init atomic scratch (graph-replay safe) -------------
__global__ void stgat_init(int* __restrict__ mmax, float* __restrict__ den,
                           float* __restrict__ gat_out) {
    int gid = blockIdx.x * 256 + threadIdx.x;
    const int natt = T_STEPS * NN * HEADS;       // 288000
    const int ngat = NN * T_STEPS * HID;         // 1152000
    if (gid < natt) { mmax[gid] = (int)0x80000000; den[gid] = 0.0f; }
    if (gid < ngat) gat_out[gid] = 0.0f;
}

// ---------------- 1: projection + attention coefficients --------------------
// wave per (t, n, head): h[t,n,hc] = x @ W^T ; a_s/a_d = <h, att>
__global__ __launch_bounds__(256) void stgat_proj_att(
        const float* __restrict__ X, const float* __restrict__ gat_w,
        const float* __restrict__ att_src, const float* __restrict__ att_dst,
        float* __restrict__ h, float* __restrict__ a_s, float* __restrict__ a_d) {
    int wid  = blockIdx.x * 8 + (threadIdx.x >> 5);
    int lane = threadIdx.x & 31;
    int t    = wid / (NN * HEADS);
    int rem  = wid - t * (NN * HEADS);
    int n    = rem >> 3;
    int hd   = rem & 7;
    int hc   = hd * HID + lane;

    float x0 = X[(n * T_STEPS + t) * F_INN + 0];
    float x1 = X[(n * T_STEPS + t) * F_INN + 1];
    float hv = x0 * gat_w[hc * 2 + 0] + x1 * gat_w[hc * 2 + 1];
    h[(size_t)(t * NN + n) * HC + hc] = hv;

    float ps = hv * att_src[hc];
    float pd = hv * att_dst[hc];
    for (int m = 16; m > 0; m >>= 1) {
        ps += __shfl_xor(ps, m, 32);
        pd += __shfl_xor(pd, m, 32);
    }
    if (lane == 0) {
        a_s[(t * NN + n) * HEADS + hd] = ps;
        a_d[(t * NN + n) * HEADS + hd] = pd;
    }
}

// ---------------- 2: edge logits -> segment max -----------------------------
__global__ void stgat_edge_max(const int* __restrict__ ei, int E,
                               const float* __restrict__ a_s, const float* __restrict__ a_d,
                               int* __restrict__ mmax) {
    int gid = blockIdx.x * 256 + threadIdx.x;
    int tot = (E + NN) * T_STEPS * HEADS;
    if (gid >= tot) return;
    int e = gid / (T_STEPS * HEADS);
    int r = gid - e * (T_STEPS * HEADS);
    int t = r >> 3, hd = r & 7;
    int src, dst;
    if (e < E) { src = ei[e]; dst = ei[E + e]; } else { src = dst = e - E; }
    float lg = a_s[(t * NN + src) * HEADS + hd] + a_d[(t * NN + dst) * HEADS + hd];
    float ev = lg > 0.0f ? lg : NEG_SLOPE * lg;
    atomicMax(&mmax[(t * NN + dst) * HEADS + hd], enc_f32(ev));
}

// ---------------- 3: exp(e - m) -> segment sum ------------------------------
__global__ void stgat_edge_den(const int* __restrict__ ei, int E,
                               const float* __restrict__ a_s, const float* __restrict__ a_d,
                               const int* __restrict__ mmax, float* __restrict__ den) {
    int gid = blockIdx.x * 256 + threadIdx.x;
    int tot = (E + NN) * T_STEPS * HEADS;
    if (gid >= tot) return;
    int e = gid / (T_STEPS * HEADS);
    int r = gid - e * (T_STEPS * HEADS);
    int t = r >> 3, hd = r & 7;
    int src, dst;
    if (e < E) { src = ei[e]; dst = ei[E + e]; } else { src = dst = e - E; }
    int di = (t * NN + dst) * HEADS + hd;
    float lg = a_s[(t * NN + src) * HEADS + hd] + a_d[di];
    float ev = lg > 0.0f ? lg : NEG_SLOPE * lg;
    atomicAdd(&den[di], expf(ev - dec_f32(mmax[di])));
}

// ---------------- 4: aggregate messages, folded over heads ------------------
// thread per (edge, t, c): sum_h alpha_h * h[src,h,c] / HEADS  -> gat_out[dst,t,c]
__global__ void stgat_aggregate(const int* __restrict__ ei, int E,
                                const float* __restrict__ a_s, const float* __restrict__ a_d,
                                const int* __restrict__ mmax, const float* __restrict__ den,
                                const float* __restrict__ h, float* __restrict__ gat_out) {
    int gid = blockIdx.x * 256 + threadIdx.x;
    int tot = (E + NN) * T_STEPS * HID;
    if (gid >= tot) return;
    int c  = gid & 31;
    int et = gid >> 5;
    int e  = et / T_STEPS;
    int t  = et - e * T_STEPS;
    int src, dst;
    if (e < E) { src = ei[e]; dst = ei[E + e]; } else { src = dst = e - E; }
    int bs = (t * NN + src) * HEADS;
    int bd = (t * NN + dst) * HEADS;
    const float* hrow = h + (size_t)(t * NN + src) * HC;
    float s = 0.0f;
#pragma unroll
    for (int hd = 0; hd < HEADS; ++hd) {
        float lg = a_s[bs + hd] + a_d[bd + hd];
        float ev = lg > 0.0f ? lg : NEG_SLOPE * lg;
        float alpha = expf(ev - dec_f32(mmax[bd + hd])) / den[bd + hd];
        s += alpha * hrow[hd * HID + c];
    }
    atomicAdd(&gat_out[(dst * T_STEPS + t) * HID + c], s * (1.0f / HEADS));
}

// ---------------- 5/6: persistent LSTM layer (WMMA) -------------------------
// Block = 16 node rows, 8 waves; wave w owns output columns [16w,16w+16) of the
// 128-wide gate matrix. Per step: G = x@Wih^T + h@Whh^T + b  (2x v_wmma_f32_16x16x32_f16)
// then elementwise LSTM update; cell state in registers, h in LDS (f16 A-frags).
__global__ __launch_bounds__(256) void stgat_lstm_layer(
        const float* __restrict__ xbase, int stride_n, int stride_t,
        const float* __restrict__ xbias,            // may be nullptr
        const float* __restrict__ w_ih, const float* __restrict__ w_hh,
        const float* __restrict__ b_ih, const float* __restrict__ b_hh,
        float* __restrict__ hs_out) {
    __shared__ float    G[16][4 * HID];   // 16 x 128 gate pre-activations
    __shared__ _Float16 hbuf[16][HID];    // h state, f16

    int tid  = threadIdx.x;
    int wave = tid >> 5, lane = tid & 31;
    int half = lane >> 4, rr = lane & 15;
    int m0   = blockIdx.x * 16;
    int nrow = m0 + rr; if (nrow >= NN) nrow = NN - 1;   // clamp pad rows
    int col  = wave * 16 + rr;                           // 0..127

    // B fragments (32x16 f16): lane = column, K = (v<4?0:16) + half*8 + (v&3)*2
    v16h bih, bhh;
#pragma unroll
    for (int v = 0; v < 8; ++v) {
        int k0 = ((v < 4) ? 0 : 16) + half * 8 + (v & 3) * 2;
        bih[2*v]   = (_Float16)w_ih[col * HID + k0];
        bih[2*v+1] = (_Float16)w_ih[col * HID + k0 + 1];
        bhh[2*v]   = (_Float16)w_hh[col * HID + k0];
        bhh[2*v+1] = (_Float16)w_hh[col * HID + k0 + 1];
    }
    float bias = b_ih[col] + b_hh[col];

    for (int i = tid; i < 16 * HID; i += 256) hbuf[i >> 5][i & 31] = (_Float16)0.0f;
    float creg0 = 0.0f, creg1 = 0.0f;   // cell state for items tid, tid+256
    __syncthreads();

    for (int t = 0; t < T_STEPS; ++t) {
        // A fragments (16x32 f16): lane = row, K = (v<4?0:16) + half*8 + (v&3)*2
        v16h ax, ah;
#pragma unroll
        for (int v = 0; v < 8; ++v) {
            int k0 = ((v < 4) ? 0 : 16) + half * 8 + (v & 3) * 2;
            const float* xp = xbase + (size_t)nrow * stride_n + (size_t)t * stride_t;
            float x0 = xp[k0], x1 = xp[k0 + 1];
            if (xbias) { x0 += xbias[k0]; x1 += xbias[k0 + 1]; }
            ax[2*v]   = (_Float16)x0;
            ax[2*v+1] = (_Float16)x1;
            ah[2*v]   = hbuf[rr][k0];
            ah[2*v+1] = hbuf[rr][k0 + 1];
        }
        v8f acc;
#pragma unroll
        for (int v = 0; v < 8; ++v) acc[v] = bias;
        acc = __builtin_amdgcn_wmma_f32_16x16x32_f16(false, ax, false, bih,
                                                     (short)0, acc, false, false);
        acc = __builtin_amdgcn_wmma_f32_16x16x32_f16(false, ah, false, bhh,
                                                     (short)0, acc, false, false);
        // C/D layout: VGPR v -> row v + 8*half, lane&15 -> column
#pragma unroll
        for (int v = 0; v < 8; ++v) G[v + 8 * half][col] = acc[v];
        __syncthreads();

        // elementwise LSTM update: 16 rows x 32 hid = 512 items, 2 per thread
#pragma unroll
        for (int it = 0; it < 2; ++it) {
            int item = tid + it * 256;
            int r = item >> 5, j = item & 31;
            float gi = G[r][j];
            float gf = G[r][HID + j];
            float gg = G[r][2 * HID + j];
            float go = G[r][3 * HID + j];
            float cprev = it ? creg1 : creg0;
            float cnew  = sigm_fast(gf) * cprev + sigm_fast(gi) * tanh_fast(gg);
            float hnew  = sigm_fast(go) * tanh_fast(cnew);
            if (it) creg1 = cnew; else creg0 = cnew;
            hbuf[r][j] = (_Float16)hnew;
            int n = m0 + r;
            if (n < NN) hs_out[((size_t)t * NN + n) * HID + j] = hnew;
        }
        __syncthreads();
    }
}

// ---------------- 7: final FC 32 -> 24 via WMMA -----------------------------
// Block = 16 rows, 2 waves; wave w owns output columns [16w,16w+16) (cols
// 24..31 zero-padded in B and masked on store). One 16x16x32 WMMA per tile.
__global__ __launch_bounds__(64) void stgat_fc_wmma(
        const float* __restrict__ hs1, const float* __restrict__ fc_w,
        const float* __restrict__ fc_b, float* __restrict__ out) {
    int tid  = threadIdx.x;
    int wave = tid >> 5, lane = tid & 31;
    int half = lane >> 4, rr = lane & 15;
    int m0   = blockIdx.x * 16;
    int nrow = m0 + rr; if (nrow >= NN) nrow = NN - 1;
    int col  = wave * 16 + rr;                 // 0..31, valid if < K_OUT

    // B fragment: B[k][col] = fc_w[col*HID + k], zero-pad cols >= K_OUT
    v16h b;
#pragma unroll
    for (int v = 0; v < 8; ++v) {
        int k0 = ((v < 4) ? 0 : 16) + half * 8 + (v & 3) * 2;
        float w0 = (col < K_OUT) ? fc_w[col * HID + k0]     : 0.0f;
        float w1 = (col < K_OUT) ? fc_w[col * HID + k0 + 1] : 0.0f;
        b[2*v]   = (_Float16)w0;
        b[2*v+1] = (_Float16)w1;
    }
    // A fragment: h_last rows
    const float* hl = hs1 + ((size_t)(T_STEPS - 1) * NN + nrow) * HID;
    v16h a;
#pragma unroll
    for (int v = 0; v < 8; ++v) {
        int k0 = ((v < 4) ? 0 : 16) + half * 8 + (v & 3) * 2;
        a[2*v]   = (_Float16)hl[k0];
        a[2*v+1] = (_Float16)hl[k0 + 1];
    }
    v8f acc;
    float bias = (col < K_OUT) ? fc_b[col] : 0.0f;
#pragma unroll
    for (int v = 0; v < 8; ++v) acc[v] = bias;
    acc = __builtin_amdgcn_wmma_f32_16x16x32_f16(false, a, false, b,
                                                 (short)0, acc, false, false);
#pragma unroll
    for (int v = 0; v < 8; ++v) {
        int n = m0 + v + 8 * half;
        if (n < NN && col < K_OUT) out[n * K_OUT + col] = acc[v];
    }
}

// ---------------- launcher --------------------------------------------------
extern "C" void kernel_launch(void* const* d_in, const int* in_sizes, int n_in,
                              void* d_out, int out_size, void* d_ws, size_t ws_size,
                              hipStream_t stream) {
    const float* X       = (const float*)d_in[0];
    const int*   ei      = (const int*)d_in[1];
    const float* gat_w   = (const float*)d_in[2];
    const float* att_src = (const float*)d_in[3];
    const float* att_dst = (const float*)d_in[4];
    const float* gat_b   = (const float*)d_in[5];
    const float* w_ih0   = (const float*)d_in[6];
    const float* w_hh0   = (const float*)d_in[7];
    const float* b_ih0   = (const float*)d_in[8];
    const float* b_hh0   = (const float*)d_in[9];
    const float* w_ih1   = (const float*)d_in[10];
    const float* w_hh1   = (const float*)d_in[11];
    const float* b_ih1   = (const float*)d_in[12];
    const float* b_hh1   = (const float*)d_in[13];
    const float* fc_w    = (const float*)d_in[14];
    const float* fc_b    = (const float*)d_in[15];
    float* out = (float*)d_out;

    const int E    = in_sizes[1] / 2;            // 32000
    const int Etot = E + NN;                     // + self loops

    // workspace layout (floats)
    float* ws   = (float*)d_ws;
    float* h    = ws;                                            // T*NN*256
    float* a_s  = h    + (size_t)T_STEPS * NN * HC;              // T*NN*8
    float* a_d  = a_s  + (size_t)T_STEPS * NN * HEADS;
    int*   mmax = (int*)(a_d + (size_t)T_STEPS * NN * HEADS);    // T*NN*8
    float* den  = (float*)mmax + (size_t)T_STEPS * NN * HEADS;
    float* gat_out = den     + (size_t)T_STEPS * NN * HEADS;     // NN*T*32
    float* hs0     = gat_out + (size_t)NN * T_STEPS * HID;       // T*NN*32
    float* hs1     = hs0     + (size_t)T_STEPS * NN * HID;

    // 0) init atomic scratch
    {
        int n = NN * T_STEPS * HID;
        stgat_init<<<(n + 255) / 256, 256, 0, stream>>>(mmax, den, gat_out);
    }
    // 1) projection + attention dots: exactly T*NN*HEADS waves
    {
        int waves = T_STEPS * NN * HEADS;
        stgat_proj_att<<<waves / 8, 256, 0, stream>>>(X, gat_w, att_src, att_dst,
                                                      h, a_s, a_d);
    }
    // 2) segment max
    {
        int n = Etot * T_STEPS * HEADS;
        stgat_edge_max<<<(n + 255) / 256, 256, 0, stream>>>(ei, E, a_s, a_d, mmax);
    }
    // 3) segment sum of exp
    {
        int n = Etot * T_STEPS * HEADS;
        stgat_edge_den<<<(n + 255) / 256, 256, 0, stream>>>(ei, E, a_s, a_d, mmax, den);
    }
    // 4) aggregate (head-folded)
    {
        int n = Etot * T_STEPS * HID;
        stgat_aggregate<<<(n + 255) / 256, 256, 0, stream>>>(ei, E, a_s, a_d, mmax,
                                                             den, h, gat_out);
    }
    // 5) LSTM layer 0 : x = gat_out (BN,T,32) + gat_b
    {
        int blocks = (NN + 15) / 16;
        stgat_lstm_layer<<<blocks, 256, 0, stream>>>(
            gat_out, T_STEPS * HID, HID, gat_b,
            w_ih0, w_hh0, b_ih0, b_hh0, hs0);
    }
    // 6) LSTM layer 1 : x = hs0 (T,BN,32)
    {
        int blocks = (NN + 15) / 16;
        stgat_lstm_layer<<<blocks, 256, 0, stream>>>(
            hs0, HID, NN * HID, nullptr,
            w_ih1, w_hh1, b_ih1, b_hh1, hs1);
    }
    // 7) FC head via WMMA
    {
        int blocks = (NN + 15) / 16;
        stgat_fc_wmma<<<blocks, 64, 0, stream>>>(hs1, fc_w, fc_b, out);
    }
}
